// ResidualVectorQuantizer_73452530696954
// MI455X (gfx1250) — compile-verified, matrix operand order checked by reading, not run on hardware
//
#include <hip/hip_runtime.h>

typedef __bf16 bf16;
typedef __attribute__((ext_vector_type(16))) __bf16 v16bf;
typedef __attribute__((ext_vector_type(8)))  __bf16 v8bf;
typedef __attribute__((ext_vector_type(2)))  __bf16 v2bf;
typedef __attribute__((ext_vector_type(8)))  float  v8f;

constexpr int kB  = 8;
constexpr int kD  = 512;
constexpr int kT  = 2048;
constexpr int kCB = 2048;
constexpr int kK  = 4;

constexpr int kRGroups = 2;          // row groups per block
constexpr int kNSplit  = 2;          // waves scanning the codebook per group
constexpr int kWaves   = kRGroups * kNSplit;
constexpr int kMB      = 4;          // 16-row M tiles per wave (64 rows/group)
constexpr int kNB      = 2;          // N tiles per accumulator block
constexpr int kRowsW   = 16 * kMB;   // rows per row-group
constexpr int kRowStride   = kD + 8; // bf16 units; 16B-aligned, bank-spread
constexpr int kGroupStride = kRowsW * kRowStride;

// dynamic LDS layout (bytes)
constexpr size_t kResBytes  = (size_t)kRGroups * kGroupStride * sizeof(bf16);
constexpr size_t kIdxOff    = kResBytes;                                  // final idx
constexpr size_t kPValOff   = kIdxOff  + (size_t)kRGroups * kRowsW * 4;   // partial val
constexpr size_t kPIdxOff   = kPValOff + (size_t)kRGroups * kNSplit * kRowsW * 4;
constexpr size_t kSmemBytes = kPIdxOff + (size_t)kRGroups * kNSplit * kRowsW * 4;

// ---------------------------------------------------------------------------
// Precompute: codebooks f32 -> bf16 (WMMA B operand) and 0.5*||c||^2 (f32).
// ---------------------------------------------------------------------------
__global__ void __launch_bounds__(128)
rvq_precompute(const float* __restrict__ cb, bf16* __restrict__ cb16,
               float* __restrict__ cnorm) {
  const int entry = blockIdx.x;            // 0 .. kK*kCB-1
  const float* src = cb + (size_t)entry * kD;
  bf16* dst = cb16 + (size_t)entry * kD;
  float ss = 0.0f;
  for (int d = threadIdx.x; d < kD; d += blockDim.x) {
    float v = src[d];
    dst[d] = (bf16)v;
    ss += v * v;
  }
  for (int off = 16; off > 0; off >>= 1) ss += __shfl_xor(ss, off, 32);
  __shared__ float wsum[4];
  const int lane = threadIdx.x & 31, w = threadIdx.x >> 5;
  if (lane == 0) wsum[w] = ss;
  __syncthreads();
  if (threadIdx.x == 0)
    cnorm[entry] = 0.5f * (wsum[0] + wsum[1] + wsum[2] + wsum[3]);
}

// ---------------------------------------------------------------------------
// Fused RVQ. A wave pair shares one 64-row bf16 residual group in LDS; each
// wave of the pair scans half the codebook (v_wmma_f32_16x16x32_bf16, 4x M-
// blocking), partial argmins merge through LDS. 2 groups/block, 128 blocks.
// ---------------------------------------------------------------------------
__global__ void __launch_bounds__(32 * kWaves)
rvq_main(const float* __restrict__ x, const float* __restrict__ cbf,
         const bf16* __restrict__ cb16, const float* __restrict__ cnorm,
         float* __restrict__ out) {
  extern __shared__ __align__(16) bf16 smem[];
  const int w    = threadIdx.x >> 5;
  const int lane = threadIdx.x & 31;
  const int rg   = w >> 1;                              // row group 0/1
  const int ns   = w & 1;                               // codebook half 0/1
  const int r0   = (blockIdx.x * kRGroups + rg) * kRowsW;
  const int b    = r0 / kT;
  const int t0   = r0 % kT;                             // same b for all rows
  const int nlow = lane & 15;                           // M (A) / N (B,C)
  const int hi   = lane >> 4;                           // K-group select

  bf16*  resg  = smem + (size_t)rg * kGroupStride;
  int*   sidx  = (int*)  ((char*)smem + kIdxOff)  + rg * kRowsW;
  float* spval = (float*)((char*)smem + kPValOff) + (rg * kNSplit) * kRowsW;
  int*   spidx = (int*)  ((char*)smem + kPIdxOff) + (rg * kNSplit) * kRowsW;

  // ---- load residual tile: resg[tt][d] = bf16(x[b, d, t0+tt]) -------------
  {
    const float* xb = x + (size_t)b * kD * kT + t0;
    for (int i = (ns << 5) + lane; i < kRowsW * (kD / 2); i += 64) {
      const int tt = i & (kRowsW - 1);
      const int d2 = i >> 6;                            // kRowsW == 64
      v2bf p;
      p[0] = (bf16)xb[(size_t)(2 * d2    ) * kT + tt];  // coalesced over tt
      p[1] = (bf16)xb[(size_t)(2 * d2 + 1) * kT + tt];
      *(v2bf*)(resg + (size_t)tt * kRowStride + 2 * d2) = p;
    }
  }
  __syncthreads();

  float* outq   = out;                                  // [B, D, T]
  float* outidx = out + (size_t)kB * kD * kT;           // [B, T] (as float)

  for (int stage = 0; stage < kK; ++stage) {
    const bf16*  cbs = cb16  + (size_t)stage * kCB * kD;
    const float* cns = cnorm + (size_t)stage * kCB;
    float minval[kMB][8];
    int   minidx[kMB][8];
#pragma unroll
    for (int mt = 0; mt < kMB; ++mt)
#pragma unroll
      for (int r = 0; r < 8; ++r) { minval[mt][r] = 3.0e38f; minidx[mt][r] = 0; }

    const bf16* arow0 = resg + (size_t)nlow * kRowStride;
    const int ntBeg = ns * (kCB / (16 * kNB * kNSplit));          // 32 per wave
    const int ntEnd = ntBeg + kCB / (16 * kNB * kNSplit);

    for (int nt = ntBeg; nt < ntEnd; ++nt) {
      v8f acc[kMB][kNB] = {};
      const bf16* bbase = cbs + (size_t)(nt * 16 * kNB + nlow) * kD;
      for (int kc = 0; kc < kD / 32; ++kc) {
        const int kb = kc * 32 + hi * 8;
        // B fragments for kNB n-tiles (global, L2-resident bf16 codebook)
        v16bf bfr[kNB];
#pragma unroll
        for (int u = 0; u < kNB; ++u) {
          const bf16* brow = bbase + (size_t)u * 16 * kD;
          v8bf blo = *(const v8bf*)(brow + kb);
          v8bf bhi = *(const v8bf*)(brow + kb + 16);
#pragma unroll
          for (int j = 0; j < 8; ++j) { bfr[u][j] = blo[j]; bfr[u][j + 8] = bhi[j]; }
        }
        // kMB A fragments from LDS; each feeds kNB WMMAs
#pragma unroll
        for (int mt = 0; mt < kMB; ++mt) {
          const bf16* arow = arow0 + (size_t)(mt * 16) * kRowStride;
          v8bf alo = *(const v8bf*)(arow + kb);
          v8bf ahi = *(const v8bf*)(arow + kb + 16);
          v16bf a;
#pragma unroll
          for (int j = 0; j < 8; ++j) { a[j] = alo[j]; a[j + 8] = ahi[j]; }
#pragma unroll
          for (int u = 0; u < kNB; ++u)
            acc[mt][u] = __builtin_amdgcn_wmma_f32_16x16x32_bf16(
                false, a, false, bfr[u], (short)0, acc[mt][u], false, false);
        }
      }
      // score = ||c||^2/2 - r.c  (argmin-equivalent to squared L2)
#pragma unroll
      for (int u = 0; u < kNB; ++u) {
        const int n = nt * 16 * kNB + u * 16 + nlow;
        const float cn = cns[n];
#pragma unroll
        for (int mt = 0; mt < kMB; ++mt)
#pragma unroll
          for (int r = 0; r < 8; ++r) {
            const float s = cn - acc[mt][u][r];
            if (s < minval[mt][r]) { minval[mt][r] = s; minidx[mt][r] = n; }
          }
      }
    }

    // ---- argmin across each 16-lane column group ---------------------------
#pragma unroll
    for (int mt = 0; mt < kMB; ++mt)
#pragma unroll
      for (int r = 0; r < 8; ++r)
#pragma unroll
        for (int off = 1; off < 16; off <<= 1) {
          const float ov = __shfl_xor(minval[mt][r], off, 32);
          const int   oi = __shfl_xor(minidx[mt][r], off, 32);
          if (ov < minval[mt][r] ||
              (ov == minval[mt][r] && oi < minidx[mt][r])) {
            minval[mt][r] = ov; minidx[mt][r] = oi;
          }
        }
    // lane 0 holds rows mt*16+0..7, lane 16 rows mt*16+8..15 -> partial tables
#pragma unroll
    for (int mt = 0; mt < kMB; ++mt)
#pragma unroll
      for (int r = 0; r < 8; ++r) {
        if (lane == 0) {
          spval[ns * kRowsW + mt * 16 + r] = minval[mt][r];
          spidx[ns * kRowsW + mt * 16 + r] = minidx[mt][r];
        }
        if (lane == 16) {
          spval[ns * kRowsW + mt * 16 + 8 + r] = minval[mt][r];
          spidx[ns * kRowsW + mt * 16 + 8 + r] = minidx[mt][r];
        }
      }
    __syncthreads();

    // merge the two codebook halves (split 0 always has the lower indices)
    if (ns == 0) {
      for (int m = lane; m < kRowsW; m += 32) {
        const float v0 = spval[m],          v1 = spval[kRowsW + m];
        const int   i0 = spidx[m];
        const int   i1 = spidx[kRowsW + m];
        sidx[m] = (v0 <= v1) ? i0 : i1;
      }
    }
    __syncthreads();

    if (stage == 0 && ns == 0)
      for (int m = lane; m < kRowsW; m += 32)
        outidx[r0 + m] = (float)sidx[m];

    // ---- residual[m][:] -= codebook_f32[idx][:] (rows split across pair) ---
    const float* cfs = cbf + (size_t)stage * kCB * kD;
#pragma unroll 1
    for (int m = ns * (kRowsW / 2); m < (ns + 1) * (kRowsW / 2); ++m) {
      const int idx = sidx[m];                          // LDS broadcast read
      const float2* q2 = (const float2*)(cfs + (size_t)idx * kD);
      v2bf* row = (v2bf*)(resg + (size_t)m * kRowStride);
      for (int d2 = lane; d2 < kD / 2; d2 += 32) {
        v2bf pk = row[d2];
        const float2 qq = q2[d2];
        v2bf np;
        np[0] = (bf16)((float)pk[0] - qq.x);
        np[1] = (bf16)((float)pk[1] - qq.y);
        row[d2] = np;
      }
    }
    __syncthreads();   // uniform across all waves
  }

  // ---- quantized_out = x - residual_final --------------------------------
  {
    const float* xb = x + (size_t)b * kD * kT + t0;
    float* ob = outq + (size_t)b * kD * kT + t0;
    for (int i = (ns << 5) + lane; i < kRowsW * (kD / 2); i += 64) {
      const int tt = i & (kRowsW - 1);
      const int d2 = i >> 6;
      v2bf p = *(const v2bf*)(resg + (size_t)tt * kRowStride + 2 * d2);
      const size_t o0 = (size_t)(2 * d2) * kT + tt;
      const size_t o1 = (size_t)(2 * d2 + 1) * kT + tt;
      ob[o0] = xb[o0] - (float)p[0];                    // coalesced over tt
      ob[o1] = xb[o1] - (float)p[1];
    }
  }
  if (blockIdx.x == 0 && threadIdx.x == 0)
    out[(size_t)kB * kD * kT + (size_t)kB * kT] = 0.0f; // loss = 0
}

// ---------------------------------------------------------------------------
extern "C" void kernel_launch(void* const* d_in, const int* in_sizes, int n_in,
                              void* d_out, int out_size, void* d_ws, size_t ws_size,
                              hipStream_t stream) {
  const float* x   = (const float*)d_in[0];   // [8, 512, 2048]
  const float* cbf = (const float*)d_in[1];   // [4, 2048, 512]
  float* out = (float*)d_out;

  // workspace: [ cnorm: K*CB f32 | cb16: K*CB*D bf16 ]  (~8.03 MB)
  float* cnorm = (float*)d_ws;
  bf16*  cb16  = (bf16*)((char*)d_ws + (size_t)kK * kCB * sizeof(float));

  // ~136 KB dynamic LDS: 2 blocks still co-reside in a 320 KB WGP
  (void)hipFuncSetAttribute((const void*)rvq_main,
                            hipFuncAttributeMaxDynamicSharedMemorySize,
                            (int)kSmemBytes);

  rvq_precompute<<<kK * kCB, 128, 0, stream>>>(cbf, cb16, cnorm);
  rvq_main<<<(kB * kT) / (kRGroups * kRowsW), 32 * kWaves, kSmemBytes, stream>>>(
      x, cbf, cb16, cnorm, out);
}
